// NeuralSDF_49271864819973
// MI455X (gfx1250) — matrix operand order repaired and checked by reading, not compiled, not run on hardware
//
#include <hip/hip_runtime.h>
#include <hip/hip_bf16.h>

typedef _Float16 v8h  __attribute__((ext_vector_type(8)));
typedef _Float16 v16h __attribute__((ext_vector_type(16)));
typedef float    v8f  __attribute__((ext_vector_type(8)));

#define NLVL     16
#define TSIZE    524288      // 2^19 hash table entries per level
#define HIDDEN   64
#define DIN      35          // 3 + 16*2
#define FDIM     13
#define ROWS     128         // points per block (4 wave32)

__constant__ int c_gs[NLVL] = {17,23,31,43,59,81,112,154,213,295,407,562,777,1073,1483,2049};

// softplus(100*x)/100, branchless & stable:
//   sp(x) = max(x,0) + 0.01*log(1 + exp(-100*|x|))
// exp arg <= 0 (no overflow); log arg in [1,2] (v_log_f32 accurate there).
// Lowers to v_exp_f32 + v_log_f32 single-instruction transcendentals.
__device__ inline float softplus100(float x) {
    float u = __expf(-100.0f * fabsf(x));
    return fmaxf(x, 0.0f) + 0.01f * __logf(1.0f + u);
}

// A operand: 16x32 f16 tile, rows = points. Documented layout:
// lane<16 : elems 0..7 = K ks+0..7,  elems 8..15 = K ks+16..23
// lane>=16: elems 0..7 = K ks+8..15, elems 8..15 = K ks+24..31
__device__ inline v16h load_A(const _Float16* buf, int rowbase, int ks, int m, int hi) {
    const _Float16* r = buf + (rowbase + m) * 64 + ks + hi * 8;
    v8h lo = *(const v8h*)(r);
    v8h hh = *(const v8h*)(r + 16);
    v16h a;
#pragma unroll
    for (int j = 0; j < 8; ++j) { a[j] = lo[j]; a[j + 8] = hh[j]; }
    return a;
}

// B operand: 32x16 f16 tile (K x N). Lane L holds column n = colbase+L%16; elems 0..7 = K ks+8*(L/16)+0..7, elems 8..15 = K ks+16+8*(L/16)+0..7.
__device__ inline v16h load_B(const _Float16* wb, int ks, int colbase, int ncols, int lane) {
    const _Float16* r = wb + (ks + ((lane >> 4) & 1) * 8) * ncols + colbase + (lane & 15);
    v16h b;
#pragma unroll
    for (int j = 0; j < 8; ++j) { b[j] = r[j * ncols]; b[j + 8] = r[(j + 16) * ncols]; }
    return b;
}

__device__ inline v8f wmma16(v16h a, v16h b, v8f c) {
    return __builtin_amdgcn_wmma_f32_16x16x32_f16(false, a, false, b, (short)0, c, false, false);
}

// ---- weight prep: W = g * V / ||V||_row, store as f16 W^T (B-operand layout) in ws ----
// ws layout (f16): B0 [64K x 64N] @0, B1 [64 x 64] @4096, B2 [64K x 16N] @8192
__global__ void prep_weights(const float* __restrict__ V0, const float* __restrict__ g0,
                             const float* __restrict__ V1, const float* __restrict__ g1,
                             const float* __restrict__ V2, const float* __restrict__ g2,
                             _Float16* __restrict__ wp) {
    int n = threadIdx.x;            // 64 threads, one output-neuron column each
    // layer 0: V0 is [64 x 35]
    {
        float s = 0.f;
        for (int k = 0; k < DIN; ++k) { float v = V0[n * DIN + k]; s += v * v; }
        float sc = g0[n] / sqrtf(s);
        for (int k = 0; k < 64; ++k)
            wp[k * 64 + n] = (_Float16)((k < DIN) ? sc * V0[n * DIN + k] : 0.0f);
    }
    // layer 1: V1 is [64 x 64]
    {
        float s = 0.f;
        for (int k = 0; k < 64; ++k) { float v = V1[n * 64 + k]; s += v * v; }
        float sc = g1[n] / sqrtf(s);
        for (int k = 0; k < 64; ++k)
            wp[4096 + k * 64 + n] = (_Float16)(sc * V1[n * 64 + k]);
    }
    // layer 2: V2 is [13 x 64], pad N to 16
    if (n < 16) {
        bool valid = n < FDIM;
        float sc = 0.f;
        if (valid) {
            float s = 0.f;
            for (int k = 0; k < 64; ++k) { float v = V2[n * 64 + k]; s += v * v; }
            sc = g2[n] / sqrtf(s);
        }
        for (int k = 0; k < 64; ++k)
            wp[8192 + k * 16 + n] = (_Float16)(valid ? sc * V2[n * 64 + k] : 0.0f);
    }
}

// ---- fused encode + 3-layer MLP ----
__global__ __launch_bounds__(128) void fused_sdf(
    const float* __restrict__ points, const float* __restrict__ table,
    const float* __restrict__ b0, const float* __restrict__ b1, const float* __restrict__ b2,
    const _Float16* __restrict__ wp, float* __restrict__ out, int N)
{
    __shared__ __align__(16) _Float16 act[ROWS * 64];   // activations, in-place per layer
    __shared__ __align__(16) _Float16 wsm[9216];        // B0|B1|B2 weights

    const int tid = threadIdx.x;

    // weights -> LDS (18432 B = 2304 u64)
    for (int i = tid; i < 2304; i += 128)
        ((unsigned long long*)wsm)[i] = ((const unsigned long long*)wp)[i];

    // ---------------- hash-grid encode (one point per thread) ----------------
    {
        int gp  = blockIdx.x * ROWS + tid;
        int gpc = gp < N ? gp : (N - 1);                 // clamp, keep EXEC full
        float px = points[(size_t)gpc * 3 + 0];
        float py = points[(size_t)gpc * 3 + 1];
        float pz = points[(size_t)gpc * 3 + 2];
        _Float16* frow = act + tid * 64;
        frow[0] = (_Float16)px; frow[1] = (_Float16)py; frow[2] = (_Float16)pz;

        float x0 = fminf(fmaxf((px + 1.0f) * 0.5f, 0.0f), 1.0f);
        float y0 = fminf(fmaxf((py + 1.0f) * 0.5f, 0.0f), 1.0f);
        float z0 = fminf(fmaxf((pz + 1.0f) * 0.5f, 0.0f), 1.0f);

        for (int lv = 0; lv < NLVL; ++lv) {
            int gs = c_gs[lv];
            float fx = x0 * (float)(gs - 1);
            float fy = y0 * (float)(gs - 1);
            float fz = z0 * (float)(gs - 1);
            int ix = (int)floorf(fx); ix = ix < 0 ? 0 : (ix > gs - 2 ? gs - 2 : ix);
            int iy = (int)floorf(fy); iy = iy < 0 ? 0 : (iy > gs - 2 ? gs - 2 : iy);
            int iz = (int)floorf(fz); iz = iz < 0 ? 0 : (iz > gs - 2 ? gs - 2 : iz);
            float wx = fx - (float)ix, wy = fy - (float)iy, wz = fz - (float)iz;
            bool dense = ((long long)gs * gs * gs) <= (long long)TSIZE;
            const float* tb = table + (size_t)lv * TSIZE * 2;

            float f0 = 0.0f, f1 = 0.0f;
#pragma unroll
            for (int cr = 0; cr < 8; ++cr) {
                int dx = (cr >> 2) & 1, dy = (cr >> 1) & 1, dz = cr & 1;
                unsigned cx = (unsigned)(ix + dx), cy = (unsigned)(iy + dy), cz = (unsigned)(iz + dz);
                unsigned idx;
                if (dense) idx = cx + (unsigned)gs * (cy + (unsigned)gs * cz);
                else       idx = (cx * 1u ^ cy * 2654435761u ^ cz * 805459861u) & (TSIZE - 1);
                float w = (dx ? wx : 1.0f - wx) * (dy ? wy : 1.0f - wy) * (dz ? wz : 1.0f - wz);
                const float* e = tb + (size_t)idx * 2;     // 8B gather, L2-resident
                f0 += e[0] * w;
                f1 += e[1] * w;
            }
            frow[3 + 2 * lv] = (_Float16)f0;
            frow[4 + 2 * lv] = (_Float16)f1;
        }
#pragma unroll
        for (int c = DIN; c < 64; ++c) frow[c] = (_Float16)0.0f;   // K padding
    }
    __syncthreads();

    const int lane = tid & 31;
    const int wv   = tid >> 5;
    const int m    = lane & 15;
    const int hi   = (lane >> 4) & 1;

    // --------------- layers 0 & 1: 64 outputs, softplus, in-place ---------------
    for (int layer = 0; layer < 2; ++layer) {
        const _Float16* wb   = wsm + layer * 4096;
        const float*    bias = (layer == 0) ? b0 : b1;
#pragma unroll
        for (int rti = 0; rti < 2; ++rti) {
            int rowbase = (wv * 2 + rti) * 16;            // wave owns its own 32 rows
            v16h a0 = load_A(act, rowbase, 0,  m, hi);
            v16h a1 = load_A(act, rowbase, 32, m, hi);
#pragma unroll
            for (int ct = 0; ct < 4; ++ct) {
                v16h bb0 = load_B(wb, 0,  ct * 16, 64, lane);
                v16h bb1 = load_B(wb, 32, ct * 16, 64, lane);
                v8f c = {0.f,0.f,0.f,0.f,0.f,0.f,0.f,0.f};
                c = wmma16(a0, bb0, c);
                c = wmma16(a1, bb1, c);
                float bv = bias[ct * 16 + m];
#pragma unroll
                for (int e = 0; e < 8; ++e) {
                    float v = softplus100(c[e] + bv);
                    act[(rowbase + e + 8 * hi) * 64 + ct * 16 + m] = (_Float16)v;
                }
            }
        }
    }

    // --------------- layer 2: 13 outputs (padded to 16), store to global ---------------
    {
        const _Float16* wb = wsm + 8192;
#pragma unroll
        for (int rti = 0; rti < 2; ++rti) {
            int rowbase = (wv * 2 + rti) * 16;
            v16h a0  = load_A(act, rowbase, 0,  m, hi);
            v16h a1  = load_A(act, rowbase, 32, m, hi);
            v16h bb0 = load_B(wb, 0,  0, 16, lane);
            v16h bb1 = load_B(wb, 32, 0, 16, lane);
            v8f c = {0.f,0.f,0.f,0.f,0.f,0.f,0.f,0.f};
            c = wmma16(a0, bb0, c);
            c = wmma16(a1, bb1, c);
            float bv = (m < FDIM) ? b2[m] : 0.0f;
#pragma unroll
            for (int e = 0; e < 8; ++e) {
                int rg = blockIdx.x * ROWS + rowbase + e + 8 * hi;
                float v = c[e] + bv;
                if (rg < N && m < FDIM) {
                    out[(size_t)N + (size_t)rg * FDIM + m] = v;   // h [N,13]
                    if (m == 0) out[rg] = v;                      // sdf [N,1]
                }
            }
        }
    }
}

extern "C" void kernel_launch(void* const* d_in, const int* in_sizes, int n_in,
                              void* d_out, int out_size, void* d_ws, size_t ws_size,
                              hipStream_t stream) {
    const float* points = (const float*)d_in[0];
    const float* table  = (const float*)d_in[1];
    const float* V0 = (const float*)d_in[2];
    const float* g0 = (const float*)d_in[3];
    const float* b0 = (const float*)d_in[4];
    const float* V1 = (const float*)d_in[5];
    const float* g1 = (const float*)d_in[6];
    const float* b1 = (const float*)d_in[7];
    const float* V2 = (const float*)d_in[8];
    const float* g2 = (const float*)d_in[9];
    const float* b2 = (const float*)d_in[10];
    float* out = (float*)d_out;
    int N = in_sizes[0] / 3;

    _Float16* wp = (_Float16*)d_ws;   // 9216 f16 = 18 KB scratch

    hipLaunchKernelGGL(prep_weights, dim3(1), dim3(64), 0, stream, V0, g0, V1, g1, V2, g2, wp);

    int nb = (N + ROWS - 1) / ROWS;
    hipLaunchKernelGGL(fused_sdf, dim3(nb), dim3(ROWS), 0, stream,
                       points, table, b0, b1, b2, (const _Float16*)d_ws, out, N);
    (void)n_in; (void)out_size; (void)ws_size;
}
